// PixelCNNDistribution_49340584297013
// MI455X (gfx1250) — matrix operand order repaired and checked by reading, not compiled, not run on hardware
//
#include <hip/hip_runtime.h>
#include <stdint.h>

// ---------------------------------------------------------------------------
// PixelCNN (F=160, NR=5) + discretized-mixture-of-logistics log-prob, MI455X.
// Convs: implicit GEMM via v_wmma_f32_16x16x32_f16, NHWC activations,
// double-buffered tile staging through gfx1250 async-to-LDS (ASYNCcnt) path.
// ---------------------------------------------------------------------------

#define Hdim 32
#define Wdim 32
#define Bdim 32
#define Mpix (Bdim * Hdim * Wdim)   // 32768 pixels
#define Fch  160
#define NRb  5
#define NMIXc 100

typedef _Float16 f16;
typedef _Float16 v16h __attribute__((ext_vector_type(16)));
typedef _Float16 v8h  __attribute__((ext_vector_type(8)));
typedef float    v8f  __attribute__((ext_vector_type(8)));

__device__ __forceinline__ float eluf(float v)      { return v > 0.f ? v : __expf(v) - 1.f; }
__device__ __forceinline__ float sigmoidf_(float v) { return 1.f / (1.f + __expf(-v)); }
__device__ __forceinline__ float softplusf_(float v){
  if (v >  20.f) return v;
  if (v < -20.f) return __expf(v);
  return logf(1.f + __expf(v));
}

// gfx1250 async copy: 16B global -> LDS, per active lane, tracked by ASYNCcnt.
// LDS address = low 32 bits of the generic pointer (flat-LDS aperture layout).
__device__ __forceinline__ void async_cp16(const void* g, void* l) {
  uint32_t laddr = (uint32_t)(uintptr_t)l;
  uint64_t gaddr = (uint64_t)(uintptr_t)g;
  asm volatile("global_load_async_to_lds_b128 %0, %1, off"
               :: "v"(laddr), "v"(gaddr) : "memory");
}
__device__ __forceinline__ void wait_async0() {
  asm volatile("s_wait_asynccnt 0" ::: "memory");
}

// ------------------------- xin = concat(2x-1, 1) (NHWC f16, C=4) ------------
__global__ void prep_xin_kernel(const float* __restrict__ x, f16* __restrict__ xin) {
  int m = blockIdx.x * blockDim.x + threadIdx.x;
  if (m >= Mpix) return;
  int xq = m & 31, y = (m >> 5) & 31, b = m >> 10;
#pragma unroll
  for (int c = 0; c < 3; ++c)
    xin[m * 4 + c] = (f16)(x[((b * 3 + c) * Hdim + y) * Wdim + xq] * 2.f - 1.f);
  xin[m * 4 + 3] = (f16)1.f;
}

// --------------- repack OIHW f32 -> [o][(dr*kw+dc)*I+ci] f16, rows padded ---
__global__ void repack_w_kernel(const float* __restrict__ W, f16* __restrict__ out,
                                int O, int I, int kh, int kw, int Npad) {
  int K = I * kh * kw;
  long idx = (long)blockIdx.x * blockDim.x + threadIdx.x;
  if (idx >= (long)Npad * K) return;
  int o = (int)(idx / K), k = (int)(idx % K);
  float v = 0.f;
  if (o < O) {
    int tap = k / I, ci = k % I;
    int dr = tap / kw, dc = tap % kw;
    v = W[((o * I + ci) * kh + dr) * kw + dc];
  }
  out[idx] = (f16)v;
}

// --------- direct conv for Cin=4 first layer (handles shift zeroing) --------
__global__ void conv_direct_kernel(const f16* __restrict__ act, int Cin,
                                   const float* __restrict__ W, const float* __restrict__ bias,
                                   int Cout, float* __restrict__ out,
                                   int kh, int kw, int pad_t, int pad_l,
                                   int zero_row0, int zero_col0, int accumulate) {
  long idx = (long)blockIdx.x * blockDim.x + threadIdx.x;
  if (idx >= (long)Mpix * Cout) return;
  int o = (int)(idx % Cout);
  int m = (int)(idx / Cout);
  int xq = m & 31, y = (m >> 5) & 31, b = m >> 10;
  float* dst = out + (size_t)m * Cout + o;
  if ((zero_row0 && y == 0) || (zero_col0 && xq == 0)) {
    if (!accumulate) *dst = 0.f;
    return;
  }
  float s = bias[o];
  for (int dr = 0; dr < kh; ++dr) {
    int yy = y + dr - pad_t;
    if ((unsigned)yy >= (unsigned)Hdim) continue;
    for (int dc = 0; dc < kw; ++dc) {
      int xx = xq + dc - pad_l;
      if ((unsigned)xx >= (unsigned)Wdim) continue;
      const f16* ap = act + (size_t)(b * 1024 + yy * 32 + xx) * Cin;
      const float* wp = W + (size_t)(o * Cin) * kh * kw + dr * kw + dc;
      for (int ci = 0; ci < Cin; ++ci)
        s += (float)ap[ci] * wp[(size_t)ci * kh * kw];
    }
  }
  if (accumulate) *dst += s; else *dst = s;
}

// ------------------- implicit-GEMM conv via WMMA f16 -> f32 -----------------
// act: NHWC f16 (Cin % 32 == 0). wts: [Npad][K] f16, K=(dr*kw+dc)*Cin+ci.
// out: NHWC f32 stride Cout. grid = (Mpix/256, Npad/64), block = 256.
// Double-buffered LDS; staging overlapped with WMMA via ASYNCcnt.
#define BM 256
#define BN 64
#define BK 32

__global__ __launch_bounds__(256)
void conv_wmma_kernel(const f16* __restrict__ act, int Cin,
                      const f16* __restrict__ wts,
                      const float* __restrict__ bias, int Cout,
                      float* __restrict__ out,
                      int kh, int kw, int pad_t, int pad_l, int accumulate) {
  __shared__ __align__(16) f16 sA[2][BM * BK];   // [buf][row(m)][k]
  __shared__ __align__(16) f16 sB[2][BN * BK];   // [buf][row(n)][k] (transposed)

  const int tid  = threadIdx.x;
  const int lane = tid & 31;
  const int wave = tid >> 5;
  const int wm   = wave >> 1;          // 0..3 -> 64-row strip
  const int wn   = wave & 1;           // 0..1 -> 32-col strip
  const int mBase = blockIdx.x * BM;
  const int nBase = blockIdx.y * BN;
  const int K  = Cin * kh * kw;
  const int nK = K >> 5;

  v8f acc[4][2];
#pragma unroll
  for (int i = 0; i < 4; ++i)
#pragma unroll
    for (int j = 0; j < 2; ++j)
#pragma unroll
      for (int e = 0; e < 8; ++e) acc[i][j][e] = 0.f;

  // A staging: one thread per row (32 halves = 4 x 16B chunks)
  const int am = mBase + tid;
  const int ax = am & 31, ay = (am >> 5) & 31, ab = am >> 10;
  // B staging: one 16B chunk per thread
  const int bn  = tid >> 2;
  const int bsg = tid & 3;

  const int laneHi = lane >> 4;   // 0/1
  const int lmod   = lane & 15;

  // stage K-chunk kk into LDS buffer bi (async; invalid rows zero-filled)
  auto stage = [&](int kk, int bi) {
    const int k0  = kk << 5;
    const int tap = k0 / Cin;          // Cin % 32 == 0 -> chunk within one tap
    const int ci0 = k0 - tap * Cin;
    const int dr = tap / kw, dc = tap % kw;
    {
      int yy = ay + dr - pad_t;
      int xx = ax + dc - pad_l;
      f16* ldst = sA[bi] + tid * BK;
      if ((unsigned)yy < (unsigned)Hdim && (unsigned)xx < (unsigned)Wdim) {
        const f16* src = act + ((size_t)(ab * 1024 + yy * 32 + xx)) * Cin + ci0;
#pragma unroll
        for (int seg = 0; seg < 4; ++seg)
          async_cp16(src + seg * 8, ldst + seg * 8);
      } else {
        uint4 z = {0u, 0u, 0u, 0u};
#pragma unroll
        for (int seg = 0; seg < 4; ++seg)
          *((uint4*)(ldst + seg * 8)) = z;
      }
    }
    {
      const f16* src = wts + (size_t)(nBase + bn) * K + k0 + bsg * 8;
      async_cp16(src, sB[bi] + bn * BK + bsg * 8);
      if (kk + 1 < nK)  // gfx1250 global_prefetch of the following weight chunk
        __builtin_prefetch(wts + (size_t)(nBase + bn) * K + (k0 + 32) + bsg * 8, 0, 0);
    }
  };

  stage(0, 0);
  for (int kk = 0; kk < nK; ++kk) {
    const int cur = kk & 1;
    wait_async0();        // own async copies into sA/sB[cur] done
    __syncthreads();      // everyone's copies visible; prev reads of cur^1 done
    if (kk + 1 < nK) stage(kk + 1, cur ^ 1);   // overlap DMA with WMMA below

    // ---- build fragments per ISA VGPR layout and issue 8 WMMAs ----
    v16h afr[4], bfr[2];
#pragma unroll
    for (int s = 0; s < 4; ++s) {
      int row = wm * 64 + s * 16 + lmod;
      v8h lo = *(const v8h*)(sA[cur] + row * BK + laneHi * 8);       // K 0-7 / 8-15
      v8h hi = *(const v8h*)(sA[cur] + row * BK + 16 + laneHi * 8);  // K 16-23 / 24-31
      v16h f;
#pragma unroll
      for (int h = 0; h < 8; ++h) { f[h] = lo[h]; f[h + 8] = hi[h]; }
      afr[s] = f;
    }
#pragma unroll
    for (int s = 0; s < 2; ++s) {
      int n = wn * 32 + s * 16 + lmod;
      v8h lo = *(const v8h*)(sB[cur] + n * BK + laneHi * 8);         // K 0-7 / 8-15
      v8h hi = *(const v8h*)(sB[cur] + n * BK + 16 + laneHi * 8);    // K 16-23 / 24-31
      v16h f;
#pragma unroll
      for (int h = 0; h < 8; ++h) { f[h] = lo[h]; f[h + 8] = hi[h]; }
      bfr[s] = f;
    }
#pragma unroll
    for (int i = 0; i < 4; ++i)
#pragma unroll
      for (int j = 0; j < 2; ++j)
        acc[i][j] = __builtin_amdgcn_wmma_f32_16x16x32_f16(
            false, afr[i], false, bfr[j], (short)0, acc[i][j], false, false);
  }

  // ---- store D (lane holds col n=lmod, rows vgpr + 8*laneHi) ----
#pragma unroll
  for (int i = 0; i < 4; ++i)
#pragma unroll
    for (int j = 0; j < 2; ++j) {
      int o = nBase + wn * 32 + j * 16 + lmod;
      if (o < Cout) {
        float bv = bias[o];
#pragma unroll
        for (int v = 0; v < 8; ++v) {
          int m = mBase + wm * 64 + i * 16 + laneHi * 8 + v;
          float val = acc[i][j][v] + bv;
          float* dst = out + (size_t)m * Cout + o;
          if (accumulate) *dst += val; else *dst = val;
        }
      }
    }
}

// ----------------- celu: F-ch f32 -> 2F-ch f16 (elu(concat(z,-z))) ----------
__global__ void celu_kernel(const float* __restrict__ in, f16* __restrict__ out, int Cf) {
  long idx = (long)blockIdx.x * blockDim.x + threadIdx.x;
  int C2 = 2 * Cf;
  if (idx >= (long)Mpix * C2) return;
  int c = (int)(idx % C2);
  long m = idx / C2;
  float v = (c < Cf) ? in[m * Cf + c] : -in[m * Cf + (c - Cf)];
  out[idx] = (f16)eluf(v);
}

// ----------------- elu: F-ch f32 -> F-ch f16 --------------------------------
__global__ void elu_kernel(const float* __restrict__ in, f16* __restrict__ out) {
  long idx = (long)blockIdx.x * blockDim.x + threadIdx.x;
  if (idx >= (long)Mpix * Fch) return;
  out[idx] = (f16)eluf(in[idx]);
}

// ----------------- gated residual: u += a * sigmoid(g) ----------------------
__global__ void gate_kernel(float* __restrict__ u, const float* __restrict__ c2) {
  long idx = (long)blockIdx.x * blockDim.x + threadIdx.x;
  if (idx >= (long)Mpix * Fch) return;
  int f = (int)(idx % Fch);
  long m = idx / Fch;
  float a = c2[m * 2 * Fch + f];
  float g = c2[m * 2 * Fch + Fch + f];
  u[idx] += a * sigmoidf_(g);
}

// ----------------- DMLL log-prob per pixel (block=128, grid=Mpix) -----------
__global__ __launch_bounds__(128)
void dmll_kernel(const float* __restrict__ x,       // NCHW (32,3,32,32)
                 const float* __restrict__ logits,  // [Mpix][1000] NHWC
                 float* __restrict__ lp_buf) {
  __shared__ float red[128];
  int m = blockIdx.x, t = threadIdx.x;
  int xq = m & 31, y = (m >> 5) & 31, b = m >> 10;
  float xs0 = x[((b * 3 + 0) * Hdim + y) * Wdim + xq] * 2.f - 1.f;
  float xs1 = x[((b * 3 + 1) * Hdim + y) * Wdim + xq] * 2.f - 1.f;
  float xs2 = x[((b * 3 + 2) * Hdim + y) * Wdim + xq] * 2.f - 1.f;
  const float* L = logits + (size_t)m * 1000;

  float logit = (t < NMIXc) ? L[t] : -1e30f;
  // logsumexp over mixture logits
  red[t] = logit; __syncthreads();
  for (int s = 64; s > 0; s >>= 1) { if (t < s) red[t] = fmaxf(red[t], red[t + s]); __syncthreads(); }
  float mx = red[0]; __syncthreads();
  red[t] = (t < NMIXc) ? __expf(logit - mx) : 0.f; __syncthreads();
  for (int s = 64; s > 0; s >>= 1) { if (t < s) red[t] += red[t + s]; __syncthreads(); }
  float lse = mx + logf(red[0]); __syncthreads();

  float tot = -1e30f;
  if (t < NMIXc) {
    int k = t;
    float mean0 = L[100 + 0 * 100 + k];
    float mean1 = L[100 + 1 * 100 + k];
    float mean2 = L[100 + 2 * 100 + k];
    float ls0 = fmaxf(L[400 + 0 * 100 + k], -7.f);
    float ls1 = fmaxf(L[400 + 1 * 100 + k], -7.f);
    float ls2 = fmaxf(L[400 + 2 * 100 + k], -7.f);
    float c0 = tanhf(L[700 + 0 * 100 + k]);
    float c1 = tanhf(L[700 + 1 * 100 + k]);
    float c2 = tanhf(L[700 + 2 * 100 + k]);
    float mu[3]  = {mean0, mean1 + c0 * xs0, mean2 + c1 * xs0 + c2 * xs1};
    float lsv[3] = {ls0, ls1, ls2};
    float xv[3]  = {xs0, xs1, xs2};
    float slp = 0.f;
#pragma unroll
    for (int ch = 0; ch < 3; ++ch) {
      float centered = xv[ch] - mu[ch];
      float inv_s = __expf(-lsv[ch]);
      float plus_in = inv_s * (centered + 1.f / 255.f);
      float min_in  = inv_s * (centered - 1.f / 255.f);
      float cdf_delta = sigmoidf_(plus_in) - sigmoidf_(min_in);
      float log_cdf_plus = plus_in - softplusf_(plus_in);
      float log_1m = -softplusf_(min_in);
      float mid = inv_s * centered;
      float log_pdf_mid = mid - lsv[ch] - 2.f * softplusf_(mid);
      float lp;
      if      (xv[ch] < -0.999f) lp = log_cdf_plus;
      else if (xv[ch] >  0.999f) lp = log_1m;
      else if (cdf_delta > 1e-5f) lp = logf(fmaxf(cdf_delta, 1e-12f));
      else                        lp = log_pdf_mid - 4.8481164f;   // log(127.5)
      slp += lp;
    }
    tot = slp + logit - lse;
  }
  // logsumexp over mixtures
  red[t] = tot; __syncthreads();
  for (int s = 64; s > 0; s >>= 1) { if (t < s) red[t] = fmaxf(red[t], red[t + s]); __syncthreads(); }
  float mx2 = red[0]; __syncthreads();
  red[t] = (t < NMIXc) ? __expf(tot - mx2) : 0.f; __syncthreads();
  for (int s = 64; s > 0; s >>= 1) { if (t < s) red[t] += red[t + s]; __syncthreads(); }
  if (t == 0) lp_buf[m] = mx2 + logf(red[0]);
}

// ----------------- fixed-order per-batch sum over 1024 pixels ---------------
__global__ __launch_bounds__(256)
void reduce_lp_kernel(const float* __restrict__ lp_buf, float* __restrict__ out) {
  __shared__ float red[256];
  int b = blockIdx.x, t = threadIdx.x;
  const float* p = lp_buf + (size_t)b * 1024;
  red[t] = p[t] + p[256 + t] + p[512 + t] + p[768 + t];
  __syncthreads();
  for (int s = 128; s > 0; s >>= 1) { if (t < s) red[t] += red[t + s]; __syncthreads(); }
  if (t == 0) out[b] = red[0];
}

// ---------------------------------------------------------------------------
extern "C" void kernel_launch(void* const* d_in, const int* in_sizes, int n_in,
                              void* d_out, int out_size, void* d_ws, size_t ws_size,
                              hipStream_t stream) {
  (void)in_sizes; (void)n_in; (void)out_size; (void)ws_size;
  const float* x      = (const float*)d_in[0];
  const float* Wui    = (const float*)d_in[1];  const float* bui   = (const float*)d_in[2];
  const float* Wulds  = (const float*)d_in[3];  const float* bulds = (const float*)d_in[4];
  const float* Wuldrs = (const float*)d_in[5];  const float* buldrs= (const float*)d_in[6];
  const float* Wu1    = (const float*)d_in[7];  const float* bu1   = (const float*)d_in[8];
  const float* Wu2    = (const float*)d_in[9];  const float* bu2   = (const float*)d_in[10];
  const float* Wul1   = (const float*)d_in[11]; const float* bul1  = (const float*)d_in[12];
  const float* Wnin   = (const float*)d_in[13]; const float* bnin  = (const float*)d_in[14];
  const float* Wul2   = (const float*)d_in[15]; const float* bul2  = (const float*)d_in[16];
  const float* Wout   = (const float*)d_in[17]; const float* bout  = (const float*)d_in[18];
  float* out = (float*)d_out;

  // ---- workspace bump allocator ----
  char* wsp = (char*)d_ws;
  auto alloc = [&](size_t bytes) -> char* {
    char* p = wsp; wsp += (bytes + 255) & ~(size_t)255; return p;
  };
  f16*   xin    = (f16*)  alloc((size_t)Mpix * 4 * 2);
  float* u      = (float*)alloc((size_t)Mpix * Fch * 4);
  float* ul     = (float*)alloc((size_t)Mpix * Fch * 4);
  f16*   bufA   = (f16*)  alloc((size_t)Mpix * 2 * Fch * 2);
  f16*   bufB   = (f16*)  alloc((size_t)Mpix * 2 * Fch * 2);
  float* t160   = (float*)alloc((size_t)Mpix * Fch * 4);
  float* t320   = (float*)alloc((size_t)Mpix * 2 * Fch * 4);
  float* logits = (float*)alloc((size_t)Mpix * 1000 * 4);
  float* lp_buf = (float*)alloc((size_t)Mpix * 4);
  // repacked weights (f16, rows padded to N-tile multiples of 64)
  f16* Wu1p  = (f16*)alloc((size_t)NRb * 192 * 1920 * 2);
  f16* Wu2p  = (f16*)alloc((size_t)NRb * 320 * 1920 * 2);
  f16* Wul1p = (f16*)alloc((size_t)NRb * 192 * 1280 * 2);
  f16* Wninp = (f16*)alloc((size_t)NRb * 192 * 320  * 2);
  f16* Wul2p = (f16*)alloc((size_t)NRb * 320 * 1280 * 2);
  f16* Woutp = (f16*)alloc((size_t)1024 * 160 * 2);

  auto repack = [&](const float* W, f16* dst, int O, int I, int kh, int kw, int Npad) {
    long tot = (long)Npad * I * kh * kw;
    repack_w_kernel<<<dim3((unsigned)((tot + 255) / 256)), 256, 0, stream>>>(
        W, dst, O, I, kh, kw, Npad);
  };
  auto conv = [&](const f16* act, int Cin, const f16* wts, const float* bias,
                  int Cout, int Npad, float* o, int kh, int kw, int pt, int pl, int accum) {
    dim3 grid(Mpix / BM, Npad / BN);
    conv_wmma_kernel<<<grid, 256, 0, stream>>>(act, Cin, wts, bias, Cout, o,
                                               kh, kw, pt, pl, accum);
  };
  auto ew = [&](long tot) { return dim3((unsigned)((tot + 255) / 256)); };

  // ---- weight repack (every call; deterministic) ----
  for (int i = 0; i < NRb; ++i) {
    repack(Wu1  + (size_t)i * 160 * 320 * 6, Wu1p  + (size_t)i * 192 * 1920, 160, 320, 2, 3, 192);
    repack(Wu2  + (size_t)i * 320 * 320 * 6, Wu2p  + (size_t)i * 320 * 1920, 320, 320, 2, 3, 320);
    repack(Wul1 + (size_t)i * 160 * 320 * 4, Wul1p + (size_t)i * 192 * 1280, 160, 320, 2, 2, 192);
    repack(Wnin + (size_t)i * 160 * 320,     Wninp + (size_t)i * 192 * 320,  160, 320, 1, 1, 192);
    repack(Wul2 + (size_t)i * 320 * 320 * 4, Wul2p + (size_t)i * 320 * 1280, 320, 320, 2, 2, 320);
  }
  repack(Wout, Woutp, 1000, 160, 1, 1, 1024);

  // ---- input prep + first-layer (shifted) convs, Cin=4 direct ----
  prep_xin_kernel<<<ew(Mpix), 256, 0, stream>>>(x, xin);
  // u  = down_shift(conv(xin, Wui, ds_pad(2,3)))        -> pt_eff=2, zero row 0
  conv_direct_kernel<<<ew((long)Mpix * Fch), 256, 0, stream>>>(
      xin, 4, Wui, bui, Fch, u, 2, 3, 2, 1, 1, 0, 0);
  // ul = down_shift(conv(xin, Wulds, ds_pad(1,3)))      -> pt_eff=1, zero row 0
  conv_direct_kernel<<<ew((long)Mpix * Fch), 256, 0, stream>>>(
      xin, 4, Wulds, bulds, Fch, ul, 1, 3, 1, 1, 1, 0, 0);
  //    + right_shift(conv(xin, Wuldrs, drs_pad(2,1)))   -> pl_eff=1, zero col 0
  conv_direct_kernel<<<ew((long)Mpix * Fch), 256, 0, stream>>>(
      xin, 4, Wuldrs, buldrs, Fch, ul, 2, 1, 1, 1, 0, 1, 1);

  // ---- gated residual blocks ----
  for (int i = 0; i < NRb; ++i) {
    const f16* wu1  = Wu1p  + (size_t)i * 192 * 1920;
    const f16* wu2  = Wu2p  + (size_t)i * 320 * 1920;
    const f16* wul1 = Wul1p + (size_t)i * 192 * 1280;
    const f16* wnin = Wninp + (size_t)i * 192 * 320;
    const f16* wul2 = Wul2p + (size_t)i * 320 * 1280;

    celu_kernel<<<ew((long)Mpix * 320), 256, 0, stream>>>(u, bufA, Fch);
    conv(bufA, 320, wu1, bu1 + i * Fch, 160, 192, t160, 2, 3, 1, 1, 0);
    celu_kernel<<<ew((long)Mpix * 320), 256, 0, stream>>>(t160, bufB, Fch);
    conv(bufB, 320, wu2, bu2 + i * 2 * Fch, 320, 320, t320, 2, 3, 1, 1, 0);
    gate_kernel<<<ew((long)Mpix * Fch), 256, 0, stream>>>(u, t320);

    celu_kernel<<<ew((long)Mpix * 320), 256, 0, stream>>>(u,  bufA, Fch);  // celu(new u)
    celu_kernel<<<ew((long)Mpix * 320), 256, 0, stream>>>(ul, bufB, Fch);  // celu(ul)
    conv(bufB, 320, wul1, bul1 + i * Fch, 160, 192, t160, 2, 2, 1, 1, 0);
    conv(bufA, 320, wnin, bnin + i * Fch, 160, 192, t160, 1, 1, 0, 0, 1);  // += 1x1 nin
    celu_kernel<<<ew((long)Mpix * 320), 256, 0, stream>>>(t160, bufA, Fch);
    conv(bufA, 320, wul2, bul2 + i * 2 * Fch, 320, 320, t320, 2, 2, 1, 1, 0);
    gate_kernel<<<ew((long)Mpix * Fch), 256, 0, stream>>>(ul, t320);
  }

  // ---- output head: elu(ul) -> 1x1 conv to 1000 logits ----
  elu_kernel<<<ew((long)Mpix * Fch), 256, 0, stream>>>(ul, bufA);
  conv(bufA, 160, Woutp, bout, 1000, 1024, logits, 1, 1, 0, 0, 0);

  // ---- DMLL log-prob + per-batch reduction ----
  dmll_kernel<<<dim3(Mpix), 128, 0, stream>>>(x, logits, lp_buf);
  reduce_lp_kernel<<<dim3(Bdim), 256, 0, stream>>>(lp_buf, out);
}
